// PointNetEstimation_22900765622994
// MI455X (gfx1250) — matrix operand order repaired and checked by reading, not compiled, not run on hardware
//
#include <hip/hip_runtime.h>

typedef __attribute__((ext_vector_type(16))) _Float16 v16h;
typedef __attribute__((ext_vector_type(8)))  _Float16 v8h;
typedef __attribute__((ext_vector_type(8)))  float    v8f;

// ---------------------------------------------------------------------------
// WMMA core: accumulate one 16x16 C tile over ntk K-steps of 32.
// A comes from LDS (row-major f16, caller passes pointer at row (mt*16+m),
// element offset hi*8).  Per the CDNA5 ISA 16-bit A 16x32 layout, lane L
// (m=L&15, hi=L>>4) needs two contiguous 8-half chunks per K-step:
//   k = k0 + hi*8 + [0..7]  and  k = k0 + 16 + hi*8 + [0..7]
// B is pre-packed fragment-major: ((nt*ntk+kt)*32 + lane) * 16 halfs,
// element j of lane L holds W[k0 + (j&7) + (L>=16?8:0) + (j>=8?16:0)][n0 + (L&15)].
// Next weight tile is prefetched at WGP scope (locality 3) so it lands in the
// near cache while the current WMMA executes.
// ---------------------------------------------------------------------------
__device__ __forceinline__ v8f mma_tile(const _Float16* __restrict__ arow,
                                        int ntk,
                                        const v16h* __restrict__ wtile) {
  v8f acc = {};
  for (int kt = 0; kt < ntk; ++kt) {
    if (kt + 1 < ntk) __builtin_prefetch((const void*)(wtile + (kt + 1) * 32), 0, 3);
    v8h lo = *(const v8h*)(arow + kt * 32);
    v8h hp = *(const v8h*)(arow + kt * 32 + 16);
    v16h a;
#pragma unroll
    for (int j = 0; j < 8; ++j) { a[j] = lo[j]; a[j + 8] = hp[j]; }
    v16h b = wtile[kt * 32];
    acc = __builtin_amdgcn_wmma_f32_16x16x32_f16(false, a, false, b,
                                                 (short)0, acc, false, false);
  }
  return acc;
}

// One fused layer inside the grouper: LDS in -> WMMA -> scale/bias/relu -> LDS out.
// LDS row stride fixed at 256 halfs.
__device__ __forceinline__ void layer_ld(const _Float16* __restrict__ in,
                                         _Float16* __restrict__ out,
                                         int rows, int cinpad, int coutpad,
                                         const _Float16* __restrict__ Wpk,
                                         const float* __restrict__ s,
                                         const float* __restrict__ t,
                                         int lane, int wave) {
  const int ntk = cinpad >> 5;
  const int ntn = coutpad >> 4;
  const int ntm = (rows + 15) >> 4;
  const int m  = lane & 15;
  const int hi = lane >> 4;
  for (int nt = wave; nt < ntn; nt += 4) {
    const v16h* wtile = (const v16h*)Wpk + (size_t)(nt * ntk) * 32 + lane;
    const int n = nt * 16 + m;
    const float sc = s[n], tb = t[n];
    for (int mt = 0; mt < ntm; ++mt) {
      const _Float16* arow = in + (mt * 16 + m) * 256 + hi * 8;
      v8f acc = mma_tile(arow, ntk, wtile);
      const int rb = mt * 16 + hi * 8;
#pragma unroll
      for (int v = 0; v < 8; ++v) {
        int r = rb + v;
        if (r < rows) {
          float y = fmaxf(acc[v] * sc + tb, 0.f);
          out[r * 256 + n] = (_Float16)y;
        }
      }
    }
  }
}

// ---------------------------------------------------------------------------
// pts (B,3,N) -> xyz (B,N,3)
// ---------------------------------------------------------------------------
__global__ void transpose_kernel(const float* __restrict__ pts,
                                 float* __restrict__ xyz, int B, int N) {
  int i = blockIdx.x * blockDim.x + threadIdx.x;
  if (i >= B * N) return;
  int b = i / N, n = i - b * N;
  const float* p = pts + (size_t)b * 3 * N + n;
  float* o = xyz + (size_t)i * 3;
  o[0] = p[0]; o[1] = p[(size_t)N]; o[2] = p[(size_t)2 * N];
}

// ---------------------------------------------------------------------------
// Farthest point sampling: one block per batch.  Matches reference semantics:
// idx[it] = entering "far"; dist = running min of squared distances; argmax
// with first-index tie-break.  new_xyz written directly.
// ---------------------------------------------------------------------------
__global__ void __launch_bounds__(256)
fps_kernel(const float* __restrict__ xyz, int N, int m,
           float* __restrict__ newxyz) {
  __shared__ float dist[2048];
  __shared__ float rv[8];
  __shared__ int   ri[8];
  __shared__ int   sFar;
  const int tid = threadIdx.x, lane = tid & 31, wave = tid >> 5;
  const int b = blockIdx.x;
  const float* X = xyz + (size_t)b * N * 3;
  for (int i = tid; i < N; i += 256) dist[i] = 1e10f;
  __syncthreads();
  int far = 0;
  for (int it = 0; it < m; ++it) {
    float cx = X[far * 3 + 0], cy = X[far * 3 + 1], cz = X[far * 3 + 2];
    if (tid == 0) {
      float* o = newxyz + (size_t)(b * m + it) * 3;
      o[0] = cx; o[1] = cy; o[2] = cz;
    }
    float best = -1.f; int bi = 0x7fffffff;
    for (int i = tid; i < N; i += 256) {
      float dx = X[i * 3 + 0] - cx, dy = X[i * 3 + 1] - cy, dz = X[i * 3 + 2] - cz;
      float d = dx * dx + dy * dy + dz * dz;
      float dd = fminf(dist[i], d);
      dist[i] = dd;
      if (dd > best || (dd == best && i < bi)) { best = dd; bi = i; }
    }
#pragma unroll
    for (int off = 16; off > 0; off >>= 1) {
      float ov = __shfl_xor(best, off, 32);
      int   oi = __shfl_xor(bi, off, 32);
      if (ov > best || (ov == best && oi < bi)) { best = ov; bi = oi; }
    }
    if (lane == 0) { rv[wave] = best; ri[wave] = bi; }
    __syncthreads();
    if (tid == 0) {
      float bv = rv[0]; int bj = ri[0];
      for (int w = 1; w < 8; ++w)
        if (rv[w] > bv || (rv[w] == bv && ri[w] < bj)) { bv = rv[w]; bj = ri[w]; }
      sFar = bj;
    }
    __syncthreads();
    far = sFar;
  }
}

// ---------------------------------------------------------------------------
// Fused grouper: one block (128 thr = 4 waves) per (batch,center).
// Ball-query (first K in-range ascending, pad with first) -> build f16 A tile
// in LDS -> 3x (WMMA GEMM + BN + ReLU) ping-pong in LDS -> max over K rows ->
// write slice of the concat buffer.
// ---------------------------------------------------------------------------
__global__ void __launch_bounds__(128)
grouper_kernel(const float* __restrict__ xyz, int N,
               const _Float16* __restrict__ feats, int featC,
               const float* __restrict__ newxyz, int NP,
               float r2, int K,
               const _Float16* W1, const float* s1, const float* t1, int cin0pad, int c1,
               const _Float16* W2, const float* s2, const float* t2, int c2,
               const _Float16* W3, const float* s3, const float* t3, int c3,
               _Float16* __restrict__ outCat, int catW, int colOfs) {
  __shared__ __attribute__((aligned(64))) _Float16 bufA[32 * 256];
  __shared__ __attribute__((aligned(64))) _Float16 bufB[32 * 256];
  __shared__ int sIdx[32];
  __shared__ int sCnt;

  const int tid = threadIdx.x, lane = tid & 31, wave = tid >> 5;
  const int bc = blockIdx.x;
  const int b = bc / NP;
  const float* C = newxyz + (size_t)bc * 3;
  const float cx = C[0], cy = C[1], cz = C[2];
  const float* X = xyz + (size_t)b * N * 3;

  if (wave == 0) {
    int cnt = 0;
    for (int base = 0; base < N && cnt < K; base += 32) {
      int i = base + lane;
      bool inr = false;
      if (i < N) {
        float dx = X[i * 3 + 0] - cx, dy = X[i * 3 + 1] - cy, dz = X[i * 3 + 2] - cz;
        inr = (dx * dx + dy * dy + dz * dz) <= r2;
      }
      unsigned long long mb = __ballot(inr);
      if (inr) {
        int pos = cnt + __popcll(mb & ((1ull << lane) - 1ull));
        if (pos < K) sIdx[pos] = i;
      }
      cnt += __popcll(mb);
      if (cnt > K) cnt = K;
    }
    if (lane == 0) sCnt = cnt;
  }
  __syncthreads();
  const int cnt = sCnt;
  const int padIdx = (cnt > 0) ? sIdx[0] : 0;

  const int CINP = cin0pad;
  for (int e = tid; e < K * CINP; e += 128) {
    int r = e / CINP, c = e - r * CINP;
    int gi = (r < cnt) ? sIdx[r] : padIdx;
    float val = 0.f;
    if (c < 3) val = X[gi * 3 + c] - C[c];
    else if (c < 3 + featC) val = (float)feats[((size_t)b * N + gi) * featC + (c - 3)];
    bufA[r * 256 + c] = (_Float16)val;
  }
  __syncthreads();
  layer_ld(bufA, bufB, K, cin0pad, c1, W1, s1, t1, lane, wave);
  __syncthreads();
  layer_ld(bufB, bufA, K, c1, c2, W2, s2, t2, lane, wave);
  __syncthreads();
  layer_ld(bufA, bufB, K, c2, c3, W3, s3, t3, lane, wave);
  __syncthreads();
  for (int c = tid; c < c3; c += 128) {
    float mx = -3.0e38f;
    for (int r = 0; r < K; ++r) mx = fmaxf(mx, (float)bufB[r * 256 + c]);
    outCat[(size_t)bc * catW + colOfs + c] = (_Float16)mx;
  }
}

// ---------------------------------------------------------------------------
// Generic WMMA GEMM + BN/ReLU epilogue.  One block per 16-row M tile,
// 4 waves split the N tiles.  Optional f16 and f32 outputs.
// ---------------------------------------------------------------------------
__global__ void __launch_bounds__(128)
gemm_kernel(const _Float16* __restrict__ A, int M, int Kpad, int Ntiles,
            const _Float16* __restrict__ Wpk,
            const float* __restrict__ s, const float* __restrict__ t, int relu,
            _Float16* __restrict__ outH, int outStride,
            float* __restrict__ outF, int outW) {
  __shared__ __attribute__((aligned(64))) _Float16 tile[16 * 512];
  const int tid = threadIdx.x, lane = tid & 31, wave = tid >> 5;
  const int m0 = blockIdx.x * 16;
  const int rows = (M - m0 < 16) ? (M - m0) : 16;
  for (int e = tid; e < 16 * Kpad; e += 128) {
    int r = e / Kpad, c = e - r * Kpad;
    tile[e] = (r < rows) ? A[(size_t)(m0 + r) * Kpad + c] : (_Float16)0.f;
  }
  __syncthreads();
  const int ntk = Kpad >> 5;
  const int m = lane & 15, hi = lane >> 4;
  for (int nt = wave; nt < Ntiles; nt += 4) {
    const v16h* wtile = (const v16h*)Wpk + (size_t)(nt * ntk) * 32 + lane;
    const _Float16* arow = tile + m * Kpad + hi * 8;
    v8f acc = mma_tile(arow, ntk, wtile);
    const int n = nt * 16 + m;
    const float sc = s[n], tb = t[n];
#pragma unroll
    for (int v = 0; v < 8; ++v) {
      int r = hi * 8 + v;
      if (r < rows) {
        float y = acc[v] * sc + tb;
        if (relu) y = fmaxf(y, 0.f);
        if (outH) outH[(size_t)(m0 + r) * outStride + n] = (_Float16)y;
        if (outF && n < outW) outF[(size_t)(m0 + r) * outW + n] = y;
      }
    }
  }
}

// ---------------------------------------------------------------------------
// Pack f32 weight [cin x cout] into WMMA B-fragment order, zero padded.
// Element index = ((nt*ntk + kt)*32 + lane)*16 + j holds
// W[kt*32 + (j&7) + (lane>=16?8:0) + (j>=8?16:0)][nt*16 + (lane&15)].
// ---------------------------------------------------------------------------
__global__ void pack_w_kernel(const float* __restrict__ W, int cin, int cout,
                              int ntk, int ntn, _Float16* __restrict__ out) {
  int total = ntk * ntn * 512;
  for (int i = blockIdx.x * blockDim.x + threadIdx.x; i < total;
       i += gridDim.x * blockDim.x) {
    int j = i & 15;
    int lane = (i >> 4) & 31;
    int tilei = i >> 9;
    int nt = tilei / ntk;
    int kt = tilei - nt * ntk;
    int k = kt * 32 + (j & 7) + ((lane & 16) ? 8 : 0) + ((j & 8) ? 16 : 0);
    int n = nt * 16 + (lane & 15);
    float v = 0.f;
    if (k < cin && n < cout) v = W[(size_t)k * cout + n];
    out[i] = (_Float16)v;
  }
}

// Fold BN (+ optional linear bias) into per-channel scale/bias.
// y = relu((xW + lb - m) * g/sqrt(v+eps) + b) = relu((xW)*s + t)
__global__ void fold_bn_kernel(const float* g, const float* b, const float* m,
                               const float* v, const float* lb,
                               int cout, int coutpad, float* s, float* t) {
  int i = blockIdx.x * blockDim.x + threadIdx.x;
  if (i >= coutpad) return;
  if (i >= cout) { s[i] = 0.f; t[i] = 0.f; return; }
  float sc = 1.f, mm = 0.f, bb = 0.f;
  if (g) { sc = g[i] * rsqrtf(v[i] + 1e-5f); mm = m[i]; bb = b[i]; }
  float bias = lb ? lb[i] : 0.f;
  s[i] = sc;
  t[i] = (bias - mm) * sc + bb;
}

// Head input: [global_feat(256) | one_hot(3) | zero pad to 288] as f16.
__global__ void build_head_kernel(const _Float16* __restrict__ f2,
                                  const float* __restrict__ oh,
                                  _Float16* __restrict__ out) {
  int i = blockIdx.x * blockDim.x + threadIdx.x;
  if (i >= 64 * 288) return;
  int b = i / 288, c = i - b * 288;
  float v = 0.f;
  if (c < 256) v = (float)f2[b * 256 + c];
  else if (c < 259) v = oh[b * 3 + (c - 256)];
  out[i] = (_Float16)v;
}

// ---------------------------------------------------------------------------
// Host side
// ---------------------------------------------------------------------------
struct LHost { int wi, bi, gi, mi, vi, lbi, cin, cinpad, cout, coutpad; };

// Flattened input indices assuming: top-level dict insertion order
// (pts, one_hot_vec, params), then pytree order inside params
// (dict keys sorted, lists in order; layer dicts W,b,g,m,v).
static const LHost LT[24] = {
  // L0 grouper0 (cin 3): 3->16->16->32
  { 23, 24, 25, 26, 27, -1,   3,  32,  16,  32},
  { 28, 29, 30, 31, 32, -1,  16,  32,  16,  32},
  { 33, 34, 35, 36, 37, -1,  16,  32,  32,  32},
  // L0 grouper1: 3->32->32->64
  { 38, 39, 40, 41, 42, -1,   3,  32,  32,  32},
  { 43, 44, 45, 46, 47, -1,  32,  32,  32,  32},
  { 48, 49, 50, 51, 52, -1,  32,  32,  64,  64},
  // L1 grouper0 (cin 67): 67->64->64->128
  { 58, 59, 60, 61, 62, -1,  67,  96,  64,  64},
  { 63, 64, 65, 66, 67, -1,  64,  64,  64,  64},
  { 68, 69, 70, 71, 72, -1,  64,  64, 128, 128},
  // L1 grouper1: 67->64->96->128
  { 73, 74, 75, 76, 77, -1,  67,  96,  64,  64},
  { 78, 79, 80, 81, 82, -1,  64,  64,  96,  96},
  { 83, 84, 85, 86, 87, -1,  96,  96, 128, 128},
  // L2 grouper0 (cin 131): 131->128->128->256
  { 93, 94, 95, 96, 97, -1, 131, 160, 128, 128},
  { 98, 99,100,101,102, -1, 128, 128, 128, 128},
  {103,104,105,106,107, -1, 128, 128, 256, 256},
  // L2 grouper1: 131->128->256->256
  {108,109,110,111,112, -1, 131, 160, 128, 128},
  {113,114,115,116,117, -1, 128, 128, 256, 256},
  {118,119,120,121,122, -1, 256, 256, 256, 256},
  // agg layers: 96->64, 256->128, 512->256
  { 18, 19, 20, 21, 22, -1,  96,  96,  64,  64},
  { 53, 54, 55, 56, 57, -1, 256, 256, 128, 128},
  { 88, 89, 90, 91, 92, -1, 512, 512, 256, 256},
  // head: fc1 (+bn1), fc2 (+bn2), fc3 (no bn/relu)
  { 12,  3,  4,  5,  6, 13, 259, 288, 256, 256},
  { 14,  8,  9, 10, 11, 15, 256, 256, 128, 128},
  { 16, -1, -1, -1, -1, 17, 128, 128,  59,  64},
};

extern "C" void kernel_launch(void* const* d_in, const int* in_sizes, int n_in,
                              void* d_out, int out_size, void* d_ws, size_t ws_size,
                              hipStream_t stream) {
  (void)in_sizes; (void)n_in; (void)out_size; (void)ws_size;
  const float* pts = (const float*)d_in[0];
  const float* oh  = (const float*)d_in[1];
  const int B = 64, N0 = 2048;

  char* ws = (char*)d_ws;
  size_t off = 0;
  auto alloc = [&](size_t bytes) -> char* {
    off = (off + 255) & ~(size_t)255;
    char* p = ws + off;
    off += bytes;
    return p;
  };

  float* xyz0 = (float*)alloc((size_t)B * N0 * 3 * 4);
  float* nx0  = (float*)alloc((size_t)B * 128 * 3 * 4);
  float* nx1  = (float*)alloc((size_t)B * 64 * 3 * 4);
  float* nx2  = (float*)alloc((size_t)B * 1 * 3 * 4);
  _Float16* cat0 = (_Float16*)alloc((size_t)B * 128 * 96 * 2);
  _Float16* f0   = (_Float16*)alloc((size_t)B * 128 * 64 * 2);
  _Float16* cat1 = (_Float16*)alloc((size_t)B * 64 * 256 * 2);
  _Float16* f1   = (_Float16*)alloc((size_t)B * 64 * 128 * 2);
  _Float16* cat2 = (_Float16*)alloc((size_t)B * 512 * 2);
  _Float16* f2   = (_Float16*)alloc((size_t)B * 256 * 2);
  _Float16* hin  = (_Float16*)alloc((size_t)B * 288 * 2);
  _Float16* h1b  = (_Float16*)alloc((size_t)B * 256 * 2);
  _Float16* h2b  = (_Float16*)alloc((size_t)B * 128 * 2);

  _Float16* pk[24]; float* ls[24]; float* lt[24];
  for (int l = 0; l < 24; ++l) {
    const LHost& L = LT[l];
    pk[l] = (_Float16*)alloc((size_t)L.cinpad * L.coutpad * 2);
    ls[l] = (float*)alloc((size_t)L.coutpad * 4);
    lt[l] = (float*)alloc((size_t)L.coutpad * 4);
    int ntk = L.cinpad / 32, ntn = L.coutpad / 16;
    int total = ntk * ntn * 512;
    pack_w_kernel<<<(total + 255) / 256, 256, 0, stream>>>(
        (const float*)d_in[L.wi], L.cin, L.cout, ntk, ntn, pk[l]);
    fold_bn_kernel<<<(L.coutpad + 255) / 256, 256, 0, stream>>>(
        L.gi >= 0 ? (const float*)d_in[L.gi] : nullptr,
        L.bi >= 0 ? (const float*)d_in[L.bi] : nullptr,
        L.mi >= 0 ? (const float*)d_in[L.mi] : nullptr,
        L.vi >= 0 ? (const float*)d_in[L.vi] : nullptr,
        L.lbi >= 0 ? (const float*)d_in[L.lbi] : nullptr,
        L.cout, L.coutpad, ls[l], lt[l]);
  }

  transpose_kernel<<<(B * N0 + 255) / 256, 256, 0, stream>>>(pts, xyz0, B, N0);

  // ---- Level 0 ----
  fps_kernel<<<B, 256, 0, stream>>>(xyz0, N0, 128, nx0);
  grouper_kernel<<<B * 128, 128, 0, stream>>>(xyz0, N0, nullptr, 0, nx0, 128,
      0.04f, 16,
      pk[0], ls[0], lt[0], 32, 32,
      pk[1], ls[1], lt[1], 32,
      pk[2], ls[2], lt[2], 32,
      cat0, 96, 0);
  grouper_kernel<<<B * 128, 128, 0, stream>>>(xyz0, N0, nullptr, 0, nx0, 128,
      0.16f, 32,
      pk[3], ls[3], lt[3], 32, 32,
      pk[4], ls[4], lt[4], 32,
      pk[5], ls[5], lt[5], 64,
      cat0, 96, 32);
  gemm_kernel<<<(B * 128 + 15) / 16, 128, 0, stream>>>(
      cat0, B * 128, 96, 4, pk[18], ls[18], lt[18], 1, f0, 64, nullptr, 0);

  // ---- Level 1 ----
  fps_kernel<<<B, 256, 0, stream>>>(nx0, 128, 64, nx1);
  grouper_kernel<<<B * 64, 128, 0, stream>>>(nx0, 128, f0, 64, nx1, 64,
      0.04f, 16,
      pk[6], ls[6], lt[6], 96, 64,
      pk[7], ls[7], lt[7], 64,
      pk[8], ls[8], lt[8], 128,
      cat1, 256, 0);
  grouper_kernel<<<B * 64, 128, 0, stream>>>(nx0, 128, f0, 64, nx1, 64,
      0.16f, 32,
      pk[9],  ls[9],  lt[9],  96, 64,
      pk[10], ls[10], lt[10], 96,
      pk[11], ls[11], lt[11], 128,
      cat1, 256, 128);
  gemm_kernel<<<(B * 64 + 15) / 16, 128, 0, stream>>>(
      cat1, B * 64, 256, 8, pk[19], ls[19], lt[19], 1, f1, 128, nullptr, 0);

  // ---- Level 2 ----
  fps_kernel<<<B, 256, 0, stream>>>(nx1, 64, 1, nx2);
  grouper_kernel<<<B, 128, 0, stream>>>(nx1, 64, f1, 128, nx2, 1,
      0.04f, 16,
      pk[12], ls[12], lt[12], 160, 128,
      pk[13], ls[13], lt[13], 128,
      pk[14], ls[14], lt[14], 256,
      cat2, 512, 0);
  grouper_kernel<<<B, 128, 0, stream>>>(nx1, 64, f1, 128, nx2, 1,
      0.16f, 32,
      pk[15], ls[15], lt[15], 160, 128,
      pk[16], ls[16], lt[16], 256,
      pk[17], ls[17], lt[17], 256,
      cat2, 512, 256);
  gemm_kernel<<<(B + 15) / 16, 128, 0, stream>>>(
      cat2, B, 512, 16, pk[20], ls[20], lt[20], 1, f2, 256, nullptr, 0);

  // ---- Head ----
  build_head_kernel<<<(B * 288 + 255) / 256, 256, 0, stream>>>(f2, oh, hin);
  gemm_kernel<<<(B + 15) / 16, 128, 0, stream>>>(
      hin, B, 288, 16, pk[21], ls[21], lt[21], 1, h1b, 256, nullptr, 0);
  gemm_kernel<<<(B + 15) / 16, 128, 0, stream>>>(
      h1b, B, 256, 8, pk[22], ls[22], lt[22], 1, h2b, 128, nullptr, 0);
  gemm_kernel<<<(B + 15) / 16, 128, 0, stream>>>(
      h2b, B, 128, 4, pk[23], ls[23], lt[23], 0, nullptr, 0,
      (float*)d_out, 59);
}
